// InteractionNetEmbed_60962765800176
// MI455X (gfx1250) — compile-verified, matrix operand order checked by reading, not run on hardware
//
#include <hip/hip_runtime.h>

#define DEV __device__ __forceinline__

typedef __attribute__((ext_vector_type(16))) __bf16 v16bf;
typedef __attribute__((ext_vector_type(8)))  float  v8f;

constexpr int B_  = 32, S_ = 4096, D_ = 64, L_ = 13, NW_ = 13;
constexpr int HF_ = 128, HV_ = 128, NC_ = 2, VOCAB_ = 512;
constexpr int BS_ = B_ * S_;              // 131072 rows
constexpr int NTILES_ = BS_ / 16;         // 8192 row-tiles
constexpr int WPB_ = 8;                   // waves per block (256 threads)
constexpr int MT_  = 4;                   // row-tiles per wave (weight reuse)
constexpr int TPB_ = WPB_ * MT_;          // row-tiles per block = 32

DEV unsigned short f32_to_bf16(float f) {
  unsigned int u = __float_as_uint(f);
  u += 0x7FFFu + ((u >> 16) & 1u);        // round-to-nearest-even
  return (unsigned short)(u >> 16);
}

union FragU { v16bf v; uint4 q[2]; };

// 16-element bf16 fragment slice: K {klo..klo+7} and {klo+16..klo+23}
DEV v16bf load_frag16(const unsigned short* base, int klo) {
  FragU f;
  f.q[0] = *(const uint4*)(base + klo);
  f.q[1] = *(const uint4*)(base + klo + 16);
  return f.v;
}

DEV v8f wmma_bf16(v16bf a, v16bf b, v8f c) {
  return __builtin_amdgcn_wmma_f32_16x16x32_bf16(
      /*neg_a=*/false, a, /*neg_b=*/false, b,
      /*c_mod=*/(short)0, c, /*reuse_a=*/false, /*reuse_b=*/false);
}

DEV float gelu_erf(float x) { return 0.5f * x * (1.0f + erff(x * 0.70710678118654752f)); }

// ---------------- prep kernels (tiny, run once per launch) ----------------

__global__ void k_pack_bf16(const float* __restrict__ src,
                            unsigned short* __restrict__ dst, int n) {
  int i = blockIdx.x * blockDim.x + threadIdx.x;
  if (i < n) dst[i] = f32_to_bf16(src[i]);
}

// src: [batch][K][N] fp32  ->  dst: [batch][Npad][K] bf16 (zero-padded cols)
__global__ void k_transpose_pack(const float* __restrict__ src,
                                 unsigned short* __restrict__ dst,
                                 int K, int N, int Npad, int total) {
  int i = blockIdx.x * blockDim.x + threadIdx.x;
  if (i >= total) return;
  int kk = i % K;
  int t  = i / K;
  int n  = t % Npad;
  int b  = t / Npad;
  float v = (n < N) ? src[((size_t)b * K + kk) * N + n] : 0.0f;
  dst[i] = f32_to_bf16(v);
}

// ---------------- VNet: V = gelu(X@vW1+b1)@vW2+b2 -------------------------

__global__ __launch_bounds__(256) void k_vnet(
    const int* __restrict__ tokens,
    const unsigned short* __restrict__ embB,   // [VOCAB][64] bf16
    const unsigned short* __restrict__ w1t,    // [128][64]  bf16 (N-major)
    const float* __restrict__ b1,              // [128]
    const unsigned short* __restrict__ w2t,    // [64][128]  bf16 (N-major)
    const float* __restrict__ b2,              // [64]
    float* __restrict__ V)                     // [BS][64]
{
  __shared__ unsigned short sW1[HV_ * D_];        // 16 KB
  __shared__ unsigned short sW2[D_ * HV_];        // 16 KB
  __shared__ unsigned short sH[WPB_][16 * HF_];   // 32 KB (per-wave scratch)

  // cooperative weight staging (global -> LDS, 16B chunks)
  {
    const uint4* g1 = (const uint4*)w1t;
    uint4*       l1 = (uint4*)sW1;
    for (int i = threadIdx.x; i < (HV_ * D_) / 8; i += 256) l1[i] = g1[i];
    const uint4* g2 = (const uint4*)w2t;
    uint4*       l2 = (uint4*)sW2;
    for (int i = threadIdx.x; i < (D_ * HV_) / 8; i += 256) l2[i] = g2[i];
  }
  __syncthreads();

  const int lane = threadIdx.x & 31;
  const int wave = threadIdx.x >> 5;
  const int hiHalf = lane >> 4;
  const int kbase  = hiHalf * 8;
  const int nl = lane & 15;
  const int tile0 = blockIdx.x * TPB_ + wave * MT_;

  v8f zero = {};
  #pragma unroll
  for (int mt = 0; mt < MT_; ++mt) {
    const int rowBase = (tile0 + mt) * 16;
    const int tok = tokens[rowBase + nl];
    const unsigned short* aRow = embB + (size_t)tok * D_;

    // ---- layer 1: 16x64 @ 64x128 (A: global gather, B: LDS) ----
    v8f acc[8];
    #pragma unroll
    for (int nt = 0; nt < 8; ++nt) acc[nt] = zero;
    #pragma unroll
    for (int c = 0; c < 2; ++c) {
      v16bf a = load_frag16(aRow, kbase + c * 32);
      #pragma unroll
      for (int nt = 0; nt < 8; ++nt) {
        v16bf b = load_frag16(sW1 + (nt * 16 + nl) * D_, kbase + c * 32);
        acc[nt] = wmma_bf16(a, b, acc[nt]);
      }
    }
    #pragma unroll
    for (int nt = 0; nt < 8; ++nt) {
      const int col = nt * 16 + nl;
      const float bias = b1[col];
      #pragma unroll
      for (int i = 0; i < 8; ++i) {
        const int m = i + hiHalf * 8;
        sH[wave][m * HF_ + col] = f32_to_bf16(gelu_erf(acc[nt][i] + bias));
      }
    }
    // per-wave buffer; same-wave LDS ops complete in order -> no barrier

    // ---- layer 2: 16x128 @ 128x64 (A: LDS H, B: LDS) ----
    v8f acc2[4];
    #pragma unroll
    for (int nt = 0; nt < 4; ++nt) acc2[nt] = zero;
    #pragma unroll
    for (int c = 0; c < 4; ++c) {
      v16bf a = load_frag16(&sH[wave][nl * HF_], kbase + c * 32);
      #pragma unroll
      for (int nt = 0; nt < 4; ++nt) {
        v16bf b = load_frag16(sW2 + (nt * 16 + nl) * HV_, kbase + c * 32);
        acc2[nt] = wmma_bf16(a, b, acc2[nt]);
      }
    }
    #pragma unroll
    for (int nt = 0; nt < 4; ++nt) {
      const int col = nt * 16 + nl;
      const float bias = b2[col];
      #pragma unroll
      for (int i = 0; i < 8; ++i) {
        const int m = i + hiHalf * 8;
        V[(size_t)(rowBase + m) * D_ + col] = acc2[nt][i] + bias;
      }
    }
  }
}

// ---------------- FNet k: Fs[k] = gelu(X@fW1[k]+b1)@fW2[k]+b2 -------------

__global__ __launch_bounds__(256) void k_fnet(
    const int* __restrict__ tokens,
    const unsigned short* __restrict__ embB,
    const unsigned short* __restrict__ fW1t,   // [NW][128][64] bf16
    const float* __restrict__ fb1,             // [NW][128]
    const unsigned short* __restrict__ fW2t,   // [NW][16][128] bf16 (L padded)
    const float* __restrict__ fb2,             // [NW][13]
    float* __restrict__ Fs)                    // [NW][BS][13]
{
  __shared__ unsigned short sW1[HF_ * D_];        // 16 KB
  __shared__ unsigned short sW2[16 * HF_];        // 4 KB
  __shared__ unsigned short sH[WPB_][16 * HF_];   // 32 KB

  const int k = blockIdx.y;
  {
    const uint4* g1 = (const uint4*)(fW1t + (size_t)k * HF_ * D_);
    uint4*       l1 = (uint4*)sW1;
    for (int i = threadIdx.x; i < (HF_ * D_) / 8; i += 256) l1[i] = g1[i];
    const uint4* g2 = (const uint4*)(fW2t + (size_t)k * 16 * HF_);
    uint4*       l2 = (uint4*)sW2;
    for (int i = threadIdx.x; i < (16 * HF_) / 8; i += 256) l2[i] = g2[i];
  }
  __syncthreads();

  const int lane = threadIdx.x & 31;
  const int wave = threadIdx.x >> 5;
  const int hiHalf = lane >> 4;
  const int kbase  = hiHalf * 8;
  const int nl = lane & 15;
  const int tile0 = blockIdx.x * TPB_ + wave * MT_;
  const float* b1 = fb1 + k * HF_;

  v8f zero = {};
  #pragma unroll
  for (int mt = 0; mt < MT_; ++mt) {
    const int rowBase = (tile0 + mt) * 16;
    const int tok = tokens[rowBase + nl];
    const unsigned short* aRow = embB + (size_t)tok * D_;

    v8f acc[8];
    #pragma unroll
    for (int nt = 0; nt < 8; ++nt) acc[nt] = zero;
    #pragma unroll
    for (int c = 0; c < 2; ++c) {
      v16bf a = load_frag16(aRow, kbase + c * 32);
      #pragma unroll
      for (int nt = 0; nt < 8; ++nt) {
        v16bf b = load_frag16(sW1 + (nt * 16 + nl) * D_, kbase + c * 32);
        acc[nt] = wmma_bf16(a, b, acc[nt]);
      }
    }
    #pragma unroll
    for (int nt = 0; nt < 8; ++nt) {
      const int col = nt * 16 + nl;
      const float bias = b1[col];
      #pragma unroll
      for (int i = 0; i < 8; ++i) {
        const int m = i + hiHalf * 8;
        sH[wave][m * HF_ + col] = f32_to_bf16(gelu_erf(acc[nt][i] + bias));
      }
    }

    // 16x128 @ 128x16 (one N-tile)
    v8f acc2 = zero;
    #pragma unroll
    for (int c = 0; c < 4; ++c) {
      v16bf a = load_frag16(&sH[wave][nl * HF_], kbase + c * 32);
      v16bf b = load_frag16(sW2 + nl * HF_, kbase + c * 32);
      acc2 = wmma_bf16(a, b, acc2);
    }
    if (nl < L_) {                               // divergence only after WMMAs
      const float bias = fb2[k * L_ + nl];
      #pragma unroll
      for (int i = 0; i < 8; ++i) {
        const int m = i + hiHalf * 8;
        Fs[((size_t)k * BS_ + rowBase + m) * L_ + nl] = acc2[i] + bias;
      }
    }
  }
}

// ---------------- one chord sparse-multiply round (Z lives in L2) ---------

__global__ __launch_bounds__(256) void k_chord(
    const float* __restrict__ Fk,   // [BS][13]
    const float* __restrict__ Zin,  // [B][S][64]
    const float* __restrict__ V,    // [B][S][64]
    float* __restrict__ Zout)
{
  const int OFF[13] = {0,1,2,4,8,16,32,64,128,256,512,1024,2048};
  const int idx = blockIdx.x * blockDim.x + threadIdx.x;   // over BS*D exactly
  const int d   = idx & (D_ - 1);
  const int row = idx >> 6;            // b*S + i   (uniform per wave)
  const int b   = row >> 12;           // /4096
  const int i   = row & (S_ - 1);
  // row is constant across each wave (32 lanes cover half a 64-ch row):
  // force scalar loads of the 13 coefficients
  const int rowU = __builtin_amdgcn_readfirstlane(row);
  const float* fp = Fk + (size_t)rowU * L_;
  float f[L_];
  #pragma unroll
  for (int l = 0; l < L_; ++l) f[l] = fp[l];
  float acc = 0.0f;
  #pragma unroll
  for (int l = 0; l < L_; ++l) {
    const int j = (i + OFF[l]) & (S_ - 1);
    acc += f[l] * Zin[(((size_t)b << 12) + j) * D_ + d];
  }
  Zout[idx] = acc + V[idx];
}

// ---------------- final tiny linear: out = V.reshape(B,-1) @ finW^T + b ---

__global__ __launch_bounds__(256) void k_final(
    const float* __restrict__ V, const float* __restrict__ finW,
    const float* __restrict__ finb, float* __restrict__ out)
{
  __shared__ float red[256];
  const int b = blockIdx.x / NC_;
  const int c = blockIdx.x % NC_;
  const int n = S_ * D_;
  const float* v = V + (size_t)b * n;
  const float* w = finW + (size_t)c * n;
  float s = 0.0f;
  for (int t = threadIdx.x; t < n; t += 256) s += v[t] * w[t];
  red[threadIdx.x] = s;
  __syncthreads();
  for (int st = 128; st > 0; st >>= 1) {
    if (threadIdx.x < st) red[threadIdx.x] += red[threadIdx.x + st];
    __syncthreads();
  }
  if (threadIdx.x == 0) out[b * NC_ + c] = red[0] + finb[c];
}

// --------------------------------------------------------------------------

extern "C" void kernel_launch(void* const* d_in, const int* in_sizes, int n_in,
                              void* d_out, int out_size, void* d_ws, size_t ws_size,
                              hipStream_t stream) {
  const int*   inputs = (const int*)  d_in[0];
  const float* emb    = (const float*)d_in[1];
  const float* fW1    = (const float*)d_in[2];
  const float* fb1    = (const float*)d_in[3];
  const float* fW2    = (const float*)d_in[4];
  const float* fb2    = (const float*)d_in[5];
  const float* vW1    = (const float*)d_in[6];
  const float* vb1    = (const float*)d_in[7];
  const float* vW2    = (const float*)d_in[8];
  const float* vb2    = (const float*)d_in[9];
  const float* finW   = (const float*)d_in[10];
  const float* finb   = (const float*)d_in[11];
  float* out  = (float*)d_out;            // first B*NC floats
  float* Zout = out + B_ * NC_;           // then Z (B,S,D)

  // workspace carve-up
  char* ws = (char*)d_ws;
  size_t off = 0;
  auto carve = [&](size_t bytes) -> void* {
    void* p = ws + off;
    off = (off + bytes + 255) & ~(size_t)255;
    return p;
  };
  unsigned short* embB = (unsigned short*)carve((size_t)VOCAB_ * D_ * 2);
  unsigned short* vW1t = (unsigned short*)carve((size_t)HV_ * D_ * 2);
  unsigned short* vW2t = (unsigned short*)carve((size_t)D_ * HV_ * 2);
  unsigned short* fW1t = (unsigned short*)carve((size_t)NW_ * HF_ * D_ * 2);
  unsigned short* fW2t = (unsigned short*)carve((size_t)NW_ * 16 * HF_ * 2);
  float* Vbuf = (float*)carve((size_t)BS_ * D_ * 4);
  float* Fs   = (float*)carve((size_t)NW_ * BS_ * L_ * 4);
  float* Zws  = (float*)carve((size_t)BS_ * D_ * 4);

  // ---- prep: bf16 conversions + N-major weight transposes ----
  {
    int n = VOCAB_ * D_;
    k_pack_bf16<<<(n + 255) / 256, 256, 0, stream>>>(emb, embB, n);
  }
  k_transpose_pack<<<(HV_ * D_ + 255) / 256, 256, 0, stream>>>(
      vW1, vW1t, D_, HV_, HV_, HV_ * D_);
  k_transpose_pack<<<(D_ * HV_ + 255) / 256, 256, 0, stream>>>(
      vW2, vW2t, HV_, D_, D_, D_ * HV_);
  k_transpose_pack<<<(NW_ * HF_ * D_ + 255) / 256, 256, 0, stream>>>(
      fW1, fW1t, D_, HF_, HF_, NW_ * HF_ * D_);
  k_transpose_pack<<<(NW_ * 16 * HF_ + 255) / 256, 256, 0, stream>>>(
      fW2, fW2t, HF_, L_, 16, NW_ * 16 * HF_);

  // ---- WMMA MLPs (weights staged in LDS, 4 row-tiles per wave) ----
  k_vnet<<<NTILES_ / TPB_, 256, 0, stream>>>(inputs, embB, vW1t, vb1, vW2t, vb2, Vbuf);
  dim3 fg(NTILES_ / TPB_, NW_);
  k_fnet<<<fg, 256, 0, stream>>>(inputs, embB, fW1t, fb1, fW2t, fb2, Fs);

  // ---- 13 chord rounds; even rounds write d_out's Z region so the last
  //      (k=12, even) lands in place; odd rounds use the workspace buffer ----
  const float* zin = Vbuf;                 // Z starts as V
  for (int k = 0; k < NW_; ++k) {
    float* zo = (k & 1) ? Zws : Zout;
    k_chord<<<(BS_ * D_) / 256, 256, 0, stream>>>(
        Fs + (size_t)k * BS_ * L_, zin, Vbuf, zo);
    zin = zo;
  }

  // ---- logits from V ----
  k_final<<<B_ * NC_, 256, 0, stream>>>(Vbuf, finW, finb, out);

  (void)in_sizes; (void)n_in; (void)out_size; (void)ws_size;
}
